// UnifiedSequentialTokenizer_5248450036151
// MI455X (gfx1250) — matrix operand order repaired and checked by reading, not compiled, not run on hardware
//
#include <hip/hip_runtime.h>
#include <hip/hip_bf16.h>
#include <math.h>

#define B_    256
#define L_    200
#define H_    256
#define MAXT  256
#define SIXH  1536
#define FOURH 1024
#define NTOK  (B_ * L_)   // 51200
#define LN_EPS 1e-5f

typedef __attribute__((ext_vector_type(16))) __bf16 v16bf;
typedef __attribute__((ext_vector_type(8)))  float  v8f;
typedef __attribute__((ext_vector_type(4)))  int    i4;   // native clang vector (nontemporal-OK)

union FragU {              // 16 bf16 = 32 bytes = two 128-bit chunks
  v16bf v;
  i4    q[2];
};

__device__ __forceinline__ __bf16 f2bf(float f) {
  unsigned u = __float_as_uint(f);
  u = (u + 0x7fffu + ((u >> 16) & 1u)) >> 16;   // round-to-nearest-even
  unsigned short s = (unsigned short)u;
  return __builtin_bit_cast(__bf16, s);
}

// ---------------------------------------------------------------- weight swizzle: f32 [K,N] row-major
// -> bf16 fragment-major [kblk][ntile][lane][e] so each lane's B fragment is one
// contiguous 32-byte run. b[e] = B[kb+16*(e>=8)+(e&7)][n], kb = 8*(lane>=16), n = lane&15.
__global__ void k_swizzle_w(const float* __restrict__ src, __bf16* __restrict__ dst,
                            int K, int N) {
  int i = blockIdx.x * 256 + threadIdx.x;
  if (i >= K * N) return;
  int e    = i & 15;
  int lane = (i >> 4) & 31;
  int nb   = (i >> 9) % (N >> 4);
  int kblk = i / ((N >> 4) << 9);
  int k = kblk * 32 + ((e >> 3) << 4) + ((lane >> 4) << 3) + (e & 7);
  int n = (nb << 4) + (lane & 15);
  dst[i] = f2bf(src[(size_t)k * N + n]);
}

// ---------------------------------------------------------------- gather + layernorm -> xn (bf16)
__global__ void __launch_bounds__(256)
k_gather_ln(const int* __restrict__ ht, const int* __restrict__ pt,
            const int* __restrict__ at, const int* __restrict__ ct,
            const int* __restrict__ tg, const int* __restrict__ gi,
            const int* __restrict__ lengths,
            const float* __restrict__ tok_emb, const float* __restrict__ tg_emb,
            const float* __restrict__ g_emb,
            const float* __restrict__ gamma, const float* __restrict__ beta,
            __bf16* __restrict__ xn) {
  int tok = blockIdx.x;
  int b = tok / L_, l = tok - b * L_;
  int t = threadIdx.x;
  __bf16* row = xn + (size_t)tok * SIXH;

  if (l >= lengths[b]) {                 // masked token: dropped later; write zeros
    __bf16 z = f2bf(0.f);
#pragma unroll
    for (int j = 0; j < 6; ++j) row[j * H_ + t] = z;
    return;
  }

  int r0 = ht[tok], r1 = pt[tok], r2 = at[tok], r3 = ct[tok];
  int r4 = tg[tok]; r4 = r4 < 0 ? 0 : (r4 > 64 ? 64 : r4);
  int r5 = gi[tok];

  float v[6];
  v[0] = tok_emb[(size_t)r0 * H_ + t];
  v[1] = tok_emb[(size_t)r1 * H_ + t];
  v[2] = tok_emb[(size_t)r2 * H_ + t];
  v[3] = tok_emb[(size_t)r3 * H_ + t];
  v[4] = tg_emb[r4 * H_ + t];
  v[5] = g_emb[r5 * H_ + t];

  float s = 0.f, sq = 0.f;
#pragma unroll
  for (int j = 0; j < 6; ++j) { s += v[j]; sq += v[j] * v[j]; }

  __shared__ float rs_[256], rq_[256];
  rs_[t] = s; rq_[t] = sq; __syncthreads();
  for (int o = 128; o > 0; o >>= 1) {
    if (t < o) { rs_[t] += rs_[t + o]; rq_[t] += rq_[t + o]; }
    __syncthreads();
  }
  float mu  = rs_[0] * (1.f / SIXH);
  float var = rq_[0] * (1.f / SIXH) - mu * mu;
  float inv = rsqrtf(var + LN_EPS);
#pragma unroll
  for (int j = 0; j < 6; ++j) {
    int i = j * H_ + t;
    row[i] = f2bf((v[j] - mu) * inv * gamma[i] + beta[i]);
  }
}

// ---------------------------------------------------------------- per-batch-row merge indexing
__global__ void __launch_bounds__(256)
k_merge_index(const int* __restrict__ gi, const int* __restrict__ lengths,
              const float* __restrict__ sep, const float* __restrict__ pos,
              int* __restrict__ tokidx,
              float* __restrict__ out_merged, float* __restrict__ out_mask) {
  int b = blockIdx.x;
  int t = threadIdx.x;
  __shared__ int s_[256];
  __shared__ int flag_[256];

  int len = lengths[b];
  int sa = 0;
  if (t < L_) {
    int g  = gi[b * L_ + t];
    int gn = (t + 1 < L_) ? gi[b * L_ + t + 1] : 0;
    sa = ((t + 1) < len && g != gn) ? 1 : 0;
  }
  flag_[t] = sa; s_[t] = sa;
  __syncthreads();

  for (int off = 1; off < 256; off <<= 1) {   // Hillis-Steele inclusive scan
    int v = (t >= off) ? s_[t - off] : 0;
    __syncthreads();
    s_[t] += v;
    __syncthreads();
  }

  int nseps  = s_[255];
  int total  = len + nseps;
  int offset = MAXT - total;                  // may be negative

  if (t < L_) {
    int sb = s_[t] - sa;                      // exclusive scan
    int ti = offset + t + sb;
    tokidx[b * L_ + t] = (t < len && ti >= 0) ? ti : -1;
  }

  int mlen = total < MAXT ? total : MAXT;
  out_mask[b * MAXT + t] = (t >= MAXT - mlen) ? 1.0f : 0.0f;

  float* orow = out_merged + (size_t)b * MAXT * H_;
  for (int j = t; j < MAXT * H_; j += 256) orow[j] = 0.f;
  __syncthreads();

  for (int i = 0; i < L_; ++i) {
    if (flag_[i]) {
      int sd = offset + i + (s_[i] - 1) + 1;
      if (sd >= 0) orow[sd * H_ + t] = sep[t] + pos[sd * H_ + t];
    }
  }
}

// ---------------------------------------------------------------- GEMM1: h1 = silu(xn @ w1 + b1)
// LDS-free. Block = 32M x 1024N: 8 waves x (2 M-tiles x 8 N-tiles).
// Each B fragment load feeds two WMMAs; xn streamed NT, read from HBM once.
__global__ void __launch_bounds__(256)
k_gemm1(const __bf16* __restrict__ xn, const __bf16* __restrict__ w1s,
        const float* __restrict__ b1, __bf16* __restrict__ h1) {
  int tid = threadIdx.x, wave = tid >> 5, lane = tid & 31;
  int tokbase = blockIdx.x * 32;
  int m = lane & 15;
  int khalf = (lane >> 4) << 3;             // 0 or 8

  const __bf16* arow0 = xn + (size_t)(tokbase + m) * SIXH + khalf;
  const __bf16* arow1 = arow0 + (size_t)16 * SIXH;

  v8f acc[2][8];
#pragma unroll
  for (int mt = 0; mt < 2; ++mt)
#pragma unroll
    for (int t = 0; t < 8; ++t) acc[mt][t] = (v8f){};

#pragma unroll 1
  for (int kblk = 0; kblk < SIXH / 32; ++kblk) {
    FragU a0, a1;
    {
      const i4* pa0 = (const i4*)(arow0 + kblk * 32);
      const i4* pa1 = (const i4*)(arow1 + kblk * 32);
      a0.q[0] = __builtin_nontemporal_load(pa0);       // K: khalf .. khalf+7
      a0.q[1] = __builtin_nontemporal_load(pa0 + 2);   // K: khalf+16 .. khalf+23
      a1.q[0] = __builtin_nontemporal_load(pa1);
      a1.q[1] = __builtin_nontemporal_load(pa1 + 2);
    }
#pragma unroll
    for (int t = 0; t < 8; ++t) {
      int nt = t * 8 + wave;                // N-tile index in [0,64)
      FragU bb;
      const i4* pb = (const i4*)(w1s + ((size_t)(kblk * 64 + nt) << 9) + (lane << 4));
      bb.q[0] = pb[0];
      bb.q[1] = pb[1];
      acc[0][t] = __builtin_amdgcn_wmma_f32_16x16x32_bf16(
          false, a0.v, false, bb.v, (short)0, acc[0][t], false, false);
      acc[1][t] = __builtin_amdgcn_wmma_f32_16x16x32_bf16(
          false, a1.v, false, bb.v, (short)0, acc[1][t], false, false);
    }
  }

  int mb = (lane >> 4) * 8;                 // D: VGPR v -> M = v + 8*(lane>=16)
#pragma unroll
  for (int t = 0; t < 8; ++t) {
    int gcol = (t * 8 + wave) * 16 + (lane & 15);
    float bias = b1[gcol];
#pragma unroll
    for (int mt = 0; mt < 2; ++mt)
#pragma unroll
      for (int v = 0; v < 8; ++v) {
        int tokr = tokbase + mt * 16 + mb + v;
        float x = acc[mt][t][v] + bias;
        h1[(size_t)tokr * FOURH + gcol] = f2bf(x / (1.f + __expf(-x)));   // SiLU
      }
  }
}

// ---------------------------------------------------------------- GEMM2 + fused scatter epilogue
// Block = 32M x 256N: 8 waves x (2 M-tiles x 2 N-tiles). h1 streamed NT (single read).
__global__ void __launch_bounds__(256)
k_gemm2(const __bf16* __restrict__ h1, const __bf16* __restrict__ w2s,
        const float* __restrict__ b2, const int* __restrict__ tokidx,
        const float* __restrict__ pos, float* __restrict__ out_merged) {
  int tid = threadIdx.x, wave = tid >> 5, lane = tid & 31;
  int tokbase = blockIdx.x * 32;
  int m = lane & 15;
  int khalf = (lane >> 4) << 3;

  const __bf16* arow0 = h1 + (size_t)(tokbase + m) * FOURH + khalf;
  const __bf16* arow1 = arow0 + (size_t)16 * FOURH;

  v8f acc[2][2];
#pragma unroll
  for (int mt = 0; mt < 2; ++mt) { acc[mt][0] = (v8f){}; acc[mt][1] = (v8f){}; }

#pragma unroll 1
  for (int kblk = 0; kblk < FOURH / 32; ++kblk) {
    FragU a0, a1;
    {
      const i4* pa0 = (const i4*)(arow0 + kblk * 32);
      const i4* pa1 = (const i4*)(arow1 + kblk * 32);
      a0.q[0] = __builtin_nontemporal_load(pa0);
      a0.q[1] = __builtin_nontemporal_load(pa0 + 2);
      a1.q[0] = __builtin_nontemporal_load(pa1);
      a1.q[1] = __builtin_nontemporal_load(pa1 + 2);
    }
#pragma unroll
    for (int t = 0; t < 2; ++t) {
      int nt = t * 8 + wave;                // N-tile index in [0,16)
      FragU bb;
      const i4* pb = (const i4*)(w2s + ((size_t)(kblk * 16 + nt) << 9) + (lane << 4));
      bb.q[0] = pb[0];
      bb.q[1] = pb[1];
      acc[0][t] = __builtin_amdgcn_wmma_f32_16x16x32_bf16(
          false, a0.v, false, bb.v, (short)0, acc[0][t], false, false);
      acc[1][t] = __builtin_amdgcn_wmma_f32_16x16x32_bf16(
          false, a1.v, false, bb.v, (short)0, acc[1][t], false, false);
    }
  }

  int mb = (lane >> 4) * 8;
#pragma unroll
  for (int t = 0; t < 2; ++t) {
    int gcol = (t * 8 + wave) * 16 + (lane & 15);
    float bias = b2[gcol];
#pragma unroll
    for (int mt = 0; mt < 2; ++mt)
#pragma unroll
      for (int v = 0; v < 8; ++v) {
        int tokr = tokbase + mt * 16 + mb + v;
        int dst = tokidx[tokr];
        if (dst >= 0) {
          int b = tokr / L_;
          out_merged[((size_t)b * MAXT + dst) * H_ + gcol] =
              acc[mt][t][v] + bias + pos[dst * H_ + gcol];
        }
      }
  }
}

// ---------------------------------------------------------------- host launcher
extern "C" void kernel_launch(void* const* d_in, const int* in_sizes, int n_in,
                              void* d_out, int out_size, void* d_ws, size_t ws_size,
                              hipStream_t stream) {
  const int*   ht      = (const int*)d_in[0];
  const int*   pt      = (const int*)d_in[1];
  const int*   at      = (const int*)d_in[2];
  const int*   ct      = (const int*)d_in[3];
  const int*   tg      = (const int*)d_in[4];
  const int*   gi      = (const int*)d_in[5];
  const int*   lengths = (const int*)d_in[6];
  const float* tok_emb = (const float*)d_in[7];
  const float* gamma   = (const float*)d_in[8];
  const float* beta    = (const float*)d_in[9];
  const float* w1      = (const float*)d_in[10];
  const float* b1      = (const float*)d_in[11];
  const float* w2      = (const float*)d_in[12];
  const float* b2      = (const float*)d_in[13];
  const float* tg_emb  = (const float*)d_in[14];
  const float* g_emb   = (const float*)d_in[15];
  const float* pos     = (const float*)d_in[16];
  const float* sep     = (const float*)d_in[17];

  char* ws = (char*)d_ws;
  size_t off = 0;
  auto carve = [&](size_t bytes) -> void* {
    void* p = ws + off;
    off = (off + bytes + 255) & ~(size_t)255;
    return p;
  };
  __bf16* xn     = (__bf16*)carve((size_t)NTOK * SIXH * 2);   // 157.3 MB
  __bf16* h1     = (__bf16*)carve((size_t)NTOK * FOURH * 2);  // 104.9 MB
  __bf16* w1s    = (__bf16*)carve((size_t)SIXH * FOURH * 2);  //   3.1 MB (swizzled)
  __bf16* w2s    = (__bf16*)carve((size_t)FOURH * H_ * 2);    //   0.5 MB (swizzled)
  int*    tokidx = (int*)carve((size_t)NTOK * 4);             //   0.2 MB
  (void)ws_size; (void)in_sizes; (void)n_in; (void)out_size;

  float* out_merged = (float*)d_out;                          // [B,256,256] f32
  float* out_mask   = out_merged + (size_t)B_ * MAXT * H_;    // [B,256] f32

  k_swizzle_w<<<(SIXH * FOURH + 255) / 256, 256, 0, stream>>>(w1, w1s, SIXH, FOURH);
  k_swizzle_w<<<(FOURH * H_ + 255) / 256, 256, 0, stream>>>(w2, w2s, FOURH, H_);

  k_gather_ln<<<NTOK, 256, 0, stream>>>(ht, pt, at, ct, tg, gi, lengths,
                                        tok_emb, tg_emb, g_emb, gamma, beta, xn);

  k_merge_index<<<B_, 256, 0, stream>>>(gi, lengths, sep, pos, tokidx,
                                        out_merged, out_mask);

  k_gemm1<<<NTOK / 32, 256, 0, stream>>>(xn, w1s, b1, h1);

  k_gemm2<<<NTOK / 32, 256, 0, stream>>>(h1, w2s, b2, tokidx, pos, out_merged);
}
